// Block_78563541778642
// MI455X (gfx1250) — matrix operand order repaired and checked by reading, not compiled, not run on hardware
//
#include <hip/hip_runtime.h>
#include <hip/hip_bf16.h>
#include <cstdint>
#include <cstddef>

// ---------------------------------------------------------------------------
// Types for CDNA5 WMMA (wave32): D(f32 16x16) = A(bf16 16x32) x B(bf16 32x16) + C
// ---------------------------------------------------------------------------
typedef __attribute__((ext_vector_type(16))) __bf16 v16bf;
typedef __attribute__((ext_vector_type(8)))  float  v8f;

union Frag16 { uint4 u[2]; v16bf v; };

static __device__ inline v8f splat8(float x) {
    v8f v;
#pragma unroll
    for (int i = 0; i < 8; ++i) v[i] = x;
    return v;
}

// CDNA5 async global->LDS copy (no VGPR staging, tracked by ASYNCcnt).
// lds_off: wave-relative LDS byte offset (low 32 bits of the flat pointer).
static __device__ inline void async_copy_b128(uint32_t lds_off, const void* gaddr) {
    asm volatile("global_load_async_to_lds_b128 %0, %1, off"
                 :: "v"(lds_off), "v"(gaddr) : "memory");
}
static __device__ inline void wait_async0() {
    asm volatile("s_wait_asynccnt 0" ::: "memory");
}
static __device__ inline uint32_t lds_off_of(const void* p) {
    return (uint32_t)(size_t)p;   // flat LDS aperture: addr[31:0] == LDS offset
}

// ---------------------------------------------------------------------------
// Transpose + fp32 -> bf16 convert:  in[K][N] (row major) -> out[N][K]
// Pre-transposing weights puts GEMM B-tiles in the [N][K] layout the 16-bit
// WMMA B fragment wants (lane = N, contiguous K halves per lane).
// ---------------------------------------------------------------------------
__global__ __launch_bounds__(256) void transpose_convert_kernel(
    const float* __restrict__ in, __bf16* __restrict__ out, int K, int N)
{
    __shared__ float t[32][33];
    const int n0 = blockIdx.x * 32;
    const int k0 = blockIdx.y * 32;
    const int tx = threadIdx.x, ty = threadIdx.y;
#pragma unroll
    for (int i = 0; i < 4; ++i) {
        int k = k0 + ty + i * 8;
        t[ty + i * 8][tx] = in[(size_t)k * N + n0 + tx];
    }
    __syncthreads();
#pragma unroll
    for (int i = 0; i < 4; ++i) {
        int n = n0 + ty + i * 8;
        out[(size_t)n * K + k0 + tx] = (__bf16)t[tx][ty + i * 8];
    }
}

// ---------------------------------------------------------------------------
// LayerNorm over D=1024, one row per 128-thread block, bf16 output.
// ---------------------------------------------------------------------------
__global__ __launch_bounds__(128) void ln_kernel(
    const float* __restrict__ x, const float* __restrict__ g,
    const float* __restrict__ b, __bf16* __restrict__ out)
{
    constexpr int D = 1024;
    __shared__ float red[128];
    const int tid = threadIdx.x;
    const size_t row = blockIdx.x;
    const float* xr = x + row * D;

    float local[8];
    float s = 0.f;
#pragma unroll
    for (int i = 0; i < 8; ++i) { local[i] = xr[tid + i * 128]; s += local[i]; }
    red[tid] = s;
    for (int off = 64; off > 0; off >>= 1) {
        __syncthreads();
        if (tid < off) red[tid] += red[tid + off];
    }
    __syncthreads();
    const float mu = red[0] * (1.0f / D);
    __syncthreads();

    float s2 = 0.f;
#pragma unroll
    for (int i = 0; i < 8; ++i) { float d = local[i] - mu; s2 += d * d; }
    red[tid] = s2;
    for (int off = 64; off > 0; off >>= 1) {
        __syncthreads();
        if (tid < off) red[tid] += red[tid + off];
    }
    __syncthreads();
    const float rinv = rsqrtf(red[0] * (1.0f / D) + 1e-5f);

#pragma unroll
    for (int i = 0; i < 8; ++i) {
        int col = tid + i * 128;
        out[row * D + col] = (__bf16)((local[i] - mu) * rinv * g[col] + b[col]);
    }
}

// ---------------------------------------------------------------------------
// WMMA GEMM:  C[M][N] = A_bf16[M][K] * BT_bf16[N][K]^T + bias (+GELU) (+resid)
// 128 threads = 4 waves; block tile 128x128; each wave owns 64x64 (16 WMMA
// accumulators); BK = 32 per v_wmma_f32_16x16x32_bf16 step.
// Double-buffered LDS filled by CDNA5 async global->LDS b128 copies: zero
// staging VGPRs (round 2's register staging spilled to scratch), next tile's
// DMA overlaps the current tile's 16 WMMAs, and completion is handled by
// s_wait_asynccnt + one workgroup barrier per K-step.
// ---------------------------------------------------------------------------
__global__ __launch_bounds__(128) void gemm_bf16_wmma_kernel(
    const __bf16* __restrict__ A, const __bf16* __restrict__ BT,
    const float* __restrict__ bias, const float* __restrict__ residual,
    float* __restrict__ outF, __bf16* __restrict__ outH,
    int M, int N, int K, int act_gelu)
{
    __shared__ __align__(16) __bf16 sA[2][128][32];   // buf stride 8192 B
    __shared__ __align__(16) __bf16 sB[2][128][32];

    const int tid  = threadIdx.x;
    const int wave = tid >> 5;
    const int lane = tid & 31;
    const int lh   = lane & 15;       // N (or M) index within 16
    const int hi   = lane >> 4;       // half-wave select
    const int wm   = (wave >> 1) * 64;
    const int wn   = (wave & 1) * 64;
    const int bM   = blockIdx.y * 128;
    const int bN   = blockIdx.x * 128;

    // Staging assignment: thread owns rows r0, r0+32, r0+64, r0+96, chunk p.
    const int r0 = tid >> 2;          // 0..31
    const int p  = tid & 3;           // 16-byte chunk within a 32-half row
    const __bf16* gA[4];
    const __bf16* gB[4];
    uint32_t la[4], lb[4];            // LDS offsets in buffer 0
#pragma unroll
    for (int i = 0; i < 4; ++i) {
        gA[i] = A  + (size_t)(bM + r0 + 32 * i) * K + p * 8;
        gB[i] = BT + (size_t)(bN + r0 + 32 * i) * K + p * 8;
        la[i] = lds_off_of(&sA[0][r0 + 32 * i][p * 8]);
        lb[i] = lds_off_of(&sB[0][r0 + 32 * i][p * 8]);
    }

    v8f acc[4][4];
#pragma unroll
    for (int i = 0; i < 4; ++i)
#pragma unroll
        for (int j = 0; j < 4; ++j) acc[i][j] = splat8(0.f);

    const int alo   = hi ? 8 : 0;     // A frag: K in {lo..lo+7, lo+16..lo+23}
    const int bbase = hi ? 8 : 0;     // B frag: K in {base..base+7, base+16..base+23}

    // Prologue: async-copy K-tile 0 into buffer 0.
#pragma unroll
    for (int i = 0; i < 4; ++i) {
        async_copy_b128(la[i], gA[i]);
        async_copy_b128(lb[i], gB[i]);
        gA[i] += 32; gB[i] += 32;
    }

    int cur = 0;
    for (int k0 = 0; k0 < K; k0 += 32) {
        wait_async0();        // this wave's copies of tile k0 are in LDS
        __syncthreads();      // all waves' copies resident; prior reads done

        const bool more = (k0 + 32) < K;
        if (more) {           // DMA next tile into the other buffer
            const uint32_t bo = (uint32_t)((cur ^ 1) * 8192);
#pragma unroll
            for (int i = 0; i < 4; ++i) {
                async_copy_b128(la[i] + bo, gA[i]);
                async_copy_b128(lb[i] + bo, gB[i]);
                gA[i] += 32; gB[i] += 32;
            }
        }

        Frag16 af[4], bf[4];
#pragma unroll
        for (int mt = 0; mt < 4; ++mt) {
            int rr = wm + mt * 16 + lh;
            af[mt].u[0] = *(const uint4*)&sA[cur][rr][alo];
            af[mt].u[1] = *(const uint4*)&sA[cur][rr][alo + 16];
        }
#pragma unroll
        for (int nt = 0; nt < 4; ++nt) {
            int nn = wn + nt * 16 + lh;
            bf[nt].u[0] = *(const uint4*)&sB[cur][nn][bbase];
            bf[nt].u[1] = *(const uint4*)&sB[cur][nn][bbase + 16];
        }
#pragma unroll
        for (int mt = 0; mt < 4; ++mt)
#pragma unroll
            for (int nt = 0; nt < 4; ++nt)
                acc[mt][nt] = __builtin_amdgcn_wmma_f32_16x16x32_bf16(
                    false, af[mt].v, false, bf[nt].v,
                    (short)0, acc[mt][nt], false, false);

        cur ^= 1;
    }

    // Epilogue: bias (+GELU) (+fp32 residual), bf16 or fp32 store.
#pragma unroll
    for (int nt = 0; nt < 4; ++nt) {
        const int col = bN + wn + nt * 16 + lh;
        const float bv = bias[col];
#pragma unroll
        for (int mt = 0; mt < 4; ++mt) {
#pragma unroll
            for (int r = 0; r < 8; ++r) {
                const int row = bM + wm + mt * 16 + r + hi * 8;
                float v = acc[mt][nt][r] + bv;
                if (act_gelu) {
                    float x3 = v * v * v;
                    v = 0.5f * v *
                        (1.0f + tanhf(0.7978845608028654f * (v + 0.044715f * x3)));
                }
                if (residual) v += residual[(size_t)row * N + col];
                if (outH) outH[(size_t)row * N + col] = (__bf16)v;
                else      outF[(size_t)row * N + col] = v;
            }
        }
    }
}

// ---------------------------------------------------------------------------
// Flash attention (causal), Hd = 64, S = 1024, bf16 in/out, fp32 softmax.
// Grid: (S/64, H, B). 128 threads = 4 waves; each wave owns a 16-row q-tile.
// Score accumulators live in the WMMA C layout (elem r -> row r + 8*hi,
// lane%16 -> column); row reductions via 16-wide shfl_xor.
// ---------------------------------------------------------------------------
__global__ __launch_bounds__(128) void attn_kernel(
    const __bf16* __restrict__ qkv, __bf16* __restrict__ ctx)
{
    constexpr int S = 1024, D = 1024, LD = 3 * D, Hd = 64;
    const int qb = blockIdx.x, h = blockIdx.y, b = blockIdx.z;
    const int tid = threadIdx.x, wave = tid >> 5, lane = tid & 31;
    const int lh = lane & 15, hi = lane >> 4;
    const int lo = hi ? 8 : 0;

    __shared__ __align__(16) __bf16 sK[64][64];    // [kv][hd] == B layout [N][K]
    __shared__ __align__(16) __bf16 sVT[64][64];   // [hd][kv] == B layout for P*V
    __shared__ __align__(16) __bf16 sP[4][16][64]; // per-wave P tile [row][kv]

    // Q fragments (A layout), loaded straight from global.
    const int qrow = qb * 64 + wave * 16;
    const __bf16* qbase = qkv + (size_t)(b * S + qrow + lh) * LD + h * Hd;
    Frag16 qf[2];
#pragma unroll
    for (int kk = 0; kk < 2; ++kk) {
        qf[kk].u[0] = *(const uint4*)(qbase + kk * 32 + lo);
        qf[kk].u[1] = *(const uint4*)(qbase + kk * 32 + lo + 16);
    }

    // Hoisted per-thread staging pointers for the K/V tiles (kb = 0), advanced
    // by 64 rows per iteration.
    const __bf16* kptr[4];
    const __bf16* vptr[4];
    int srow[4], sch[4];
#pragma unroll
    for (int j = 0; j < 4; ++j) {
        int idx = tid + j * 128;
        srow[j] = idx >> 3;
        sch[j]  = idx & 7;
        kptr[j] = qkv + (size_t)(b * S + srow[j]) * LD + D     + h * Hd + sch[j] * 8;
        vptr[j] = qkv + (size_t)(b * S + srow[j]) * LD + 2 * D + h * Hd + sch[j] * 8;
    }

    v8f m = splat8(-3.0e38f), l = splat8(0.f);
    v8f O[4];
#pragma unroll
    for (int i = 0; i < 4; ++i) O[i] = splat8(0.f);

    for (int kb = 0; kb <= qb; ++kb) {            // causal: only lower blocks
        __syncthreads();
        // Cooperative load of K (as-is) and V (transposed) tiles.
#pragma unroll
        for (int j = 0; j < 4; ++j) {
            uint4 kv4 = *(const uint4*)kptr[j];
            *(uint4*)&sK[srow[j]][sch[j] * 8] = kv4;
            uint4 vv = *(const uint4*)vptr[j];
            __builtin_prefetch((const void*)(kptr[j] + 64 * LD), 0, 1);
            __builtin_prefetch((const void*)(vptr[j] + 64 * LD), 0, 1);
            kptr[j] += 64 * LD; vptr[j] += 64 * LD;
            const __bf16* ve = (const __bf16*)&vv;
#pragma unroll
            for (int jj = 0; jj < 8; ++jj) sVT[sch[j] * 8 + jj][srow[j]] = ve[jj];
        }
        __syncthreads();

        // S = (Q K^T) * 1/sqrt(Hd), causal-masked, in C layout.
        v8f st[4];
#pragma unroll
        for (int nt = 0; nt < 4; ++nt) {
            v8f acc = splat8(0.f);
#pragma unroll
            for (int kk = 0; kk < 2; ++kk) {
                Frag16 bf;
                const int n = nt * 16 + lh;
                const int base = (hi ? 8 : 0) + kk * 32;
                bf.u[0] = *(const uint4*)&sK[n][base];
                bf.u[1] = *(const uint4*)&sK[n][base + 16];
                acc = __builtin_amdgcn_wmma_f32_16x16x32_bf16(
                    false, qf[kk].v, false, bf.v, (short)0, acc, false, false);
            }
            const int kvg = kb * 64 + nt * 16 + lh;
#pragma unroll
            for (int r = 0; r < 8; ++r) {
                const int qg = qrow + r + hi * 8;
                float s = acc[r] * 0.125f;
                st[nt][r] = (kvg <= qg) ? s : -3.0e38f;
            }
        }

        // Online softmax: row max / sum over N (16 lanes of each half-wave).
        v8f mnew = m;
#pragma unroll
        for (int nt = 0; nt < 4; ++nt) {
            v8f t = st[nt];
            for (int off = 8; off >= 1; off >>= 1)
#pragma unroll
                for (int e = 0; e < 8; ++e)
                    t[e] = fmaxf(t[e], __shfl_xor(t[e], off, 16));
#pragma unroll
            for (int e = 0; e < 8; ++e) mnew[e] = fmaxf(mnew[e], t[e]);
        }
        v8f alpha;
#pragma unroll
        for (int e = 0; e < 8; ++e) alpha[e] = expf(m[e] - mnew[e]);

        v8f rowsum = splat8(0.f);
#pragma unroll
        for (int nt = 0; nt < 4; ++nt) {
#pragma unroll
            for (int e = 0; e < 8; ++e) st[nt][e] = expf(st[nt][e] - mnew[e]);
            v8f t = st[nt];
            for (int off = 8; off >= 1; off >>= 1)
#pragma unroll
                for (int e = 0; e < 8; ++e) t[e] += __shfl_xor(t[e], off, 16);
#pragma unroll
            for (int e = 0; e < 8; ++e) rowsum[e] += t[e];
        }
#pragma unroll
        for (int e = 0; e < 8; ++e) l[e] = l[e] * alpha[e] + rowsum[e];
#pragma unroll
        for (int ot = 0; ot < 4; ++ot)
#pragma unroll
            for (int e = 0; e < 8; ++e) O[ot][e] *= alpha[e];
        m = mnew;

        // Spill P to LDS (C layout -> [row][kv]), then re-read as A fragments.
#pragma unroll
        for (int nt = 0; nt < 4; ++nt)
#pragma unroll
            for (int e = 0; e < 8; ++e)
                sP[wave][e + hi * 8][nt * 16 + lh] = (__bf16)st[nt][e];
        __syncthreads();   // uniform: loop bounds identical for all 4 waves

        // O += P (16x64) * V (64x64): two K=32 WMMA steps per hd tile.
#pragma unroll
        for (int kk = 0; kk < 2; ++kk) {
            Frag16 pf;
            pf.u[0] = *(const uint4*)&sP[wave][lh][kk * 32 + lo];
            pf.u[1] = *(const uint4*)&sP[wave][lh][kk * 32 + lo + 16];
#pragma unroll
            for (int ot = 0; ot < 4; ++ot) {
                Frag16 bf;
                const int n = ot * 16 + lh;                // hd column
                const int base = (hi ? 8 : 0) + kk * 32;   // kv position
                bf.u[0] = *(const uint4*)&sVT[n][base];
                bf.u[1] = *(const uint4*)&sVT[n][base + 16];
                O[ot] = __builtin_amdgcn_wmma_f32_16x16x32_bf16(
                    false, pf.v, false, bf.v, (short)0, O[ot], false, false);
            }
        }
    }

    // Normalize and write ctx (merged heads layout [B*S][D]).
    v8f invl;
#pragma unroll
    for (int e = 0; e < 8; ++e) invl[e] = 1.0f / l[e];
#pragma unroll
    for (int ot = 0; ot < 4; ++ot)
#pragma unroll
        for (int e = 0; e < 8; ++e) {
            const int row = qrow + e + hi * 8;
            ctx[(size_t)(b * S + row) * D + h * Hd + ot * 16 + lh] =
                (__bf16)(O[ot][e] * invl[e]);
        }
}

// ---------------------------------------------------------------------------
// Host-side orchestration.
// ---------------------------------------------------------------------------
extern "C" void kernel_launch(void* const* d_in, const int* in_sizes, int n_in,
                              void* d_out, int out_size, void* d_ws, size_t ws_size,
                              hipStream_t stream) {
    (void)in_sizes; (void)n_in; (void)out_size; (void)ws_size;

    const float* x        = (const float*)d_in[0];
    const float* ln1_g    = (const float*)d_in[1];
    const float* ln1_b    = (const float*)d_in[2];
    const float* c_attn_w = (const float*)d_in[3];
    const float* c_attn_b = (const float*)d_in[4];
    const float* c_proj_w = (const float*)d_in[5];
    const float* c_proj_b = (const float*)d_in[6];
    const float* ln2_g    = (const float*)d_in[7];
    const float* ln2_b    = (const float*)d_in[8];
    const float* fc_w     = (const float*)d_in[9];
    const float* fc_b     = (const float*)d_in[10];
    const float* proj_w   = (const float*)d_in[11];
    const float* proj_b   = (const float*)d_in[12];
    float* out = (float*)d_out;

    constexpr size_t BS = 8192;            // B*S rows
    const size_t szLN1 = BS * 1024 * 2;
    const size_t szWA  = 3072ull * 1024 * 2;
    const size_t szQKV = BS * 3072 * 2;
    const size_t regA  = szLN1 + szWA + szQKV;   // reused by ff1 (64MB <= 70MB)
    const size_t szCTX = BS * 1024 * 2;
    const size_t szWP  = 1024ull * 1024 * 2;
    const size_t szH   = BS * 1024 * 4;
    const size_t szH2  = BS * 1024 * 2;
    const size_t szWFC = 4096ull * 1024 * 2;

    char* w = (char*)d_ws;
    __bf16* ln1buf  = (__bf16*)w;
    __bf16* wattnT  = (__bf16*)(w + szLN1);
    __bf16* qkv     = (__bf16*)(w + szLN1 + szWA);
    __bf16* ff1     = (__bf16*)w;                       // region A reuse
    char* pB = w + regA;
    __bf16* ctx     = (__bf16*)pB;
    __bf16* wprojT  = (__bf16*)(pB + szCTX);
    float*  hbuf    = (float*) (pB + szCTX + szWP);
    __bf16* h2buf   = (__bf16*)(pB + szCTX + szWP + szH);
    __bf16* wfcT    = (__bf16*)(pB + szCTX + szWP + szH + szH2);
    __bf16* wproj2T = (__bf16*)(pB + szCTX + szWP + szH + szH2 + szWFC);

    const dim3 tb(32, 8);
    transpose_convert_kernel<<<dim3(3072/32, 1024/32), tb, 0, stream>>>(c_attn_w, wattnT, 1024, 3072);
    transpose_convert_kernel<<<dim3(1024/32, 1024/32), tb, 0, stream>>>(c_proj_w, wprojT, 1024, 1024);
    transpose_convert_kernel<<<dim3(4096/32, 1024/32), tb, 0, stream>>>(fc_w,     wfcT,   1024, 4096);
    transpose_convert_kernel<<<dim3(1024/32, 4096/32), tb, 0, stream>>>(proj_w,   wproj2T,4096, 1024);

    // h1 = LN1(x) -> bf16
    ln_kernel<<<8192, 128, 0, stream>>>(x, ln1_g, ln1_b, ln1buf);

    // qkv = h1 @ Wattn + b  (bf16 out)
    gemm_bf16_wmma_kernel<<<dim3(3072/128, 8192/128), 128, 0, stream>>>(
        ln1buf, wattnT, c_attn_b, nullptr, nullptr, qkv, 8192, 3072, 1024, 0);

    // ctx = causal attention(q,k,v)  (bf16)
    attn_kernel<<<dim3(16, 16, 8), 128, 0, stream>>>(qkv, ctx);

    // h = x + ctx @ Wproj + b  (fp32 residual stream)
    gemm_bf16_wmma_kernel<<<dim3(1024/128, 8192/128), 128, 0, stream>>>(
        ctx, wprojT, c_proj_b, x, hbuf, nullptr, 8192, 1024, 1024, 0);

    // h2 = LN2(h) -> bf16
    ln_kernel<<<8192, 128, 0, stream>>>(hbuf, ln2_g, ln2_b, h2buf);

    // ff1 = gelu(h2 @ Wfc + b)  (bf16)
    gemm_bf16_wmma_kernel<<<dim3(4096/128, 8192/128), 128, 0, stream>>>(
        h2buf, wfcT, fc_b, nullptr, nullptr, ff1, 8192, 4096, 1024, 1);

    // out = h + ff1 @ Wproj2 + b  (fp32)
    gemm_bf16_wmma_kernel<<<dim3(1024/128, 8192/128), 128, 0, stream>>>(
        ff1, wproj2T, proj_b, hbuf, out, nullptr, 8192, 1024, 4096, 0);
}
